// FastMVSNet_singleframe_46858093199613
// MI455X (gfx1250) — matrix operand order repaired and checked by reading, not compiled, not run on hardware
//
#include <hip/hip_runtime.h>
#include <hip/hip_bf16.h>
#include <math.h>

// ---------------------------------------------------------------------------
// FastMVSNet-style forward for MI455X (gfx1250), wave32 + WMMA f16->f32.
// Implicit-GEMM convs on v_wmma_f32_16x16x32_f16; register-built fragments,
// fully branchless gathers (select-based digit carries, clamped unconditional
// addressing), constant-divisor K-decode, f16 activation storage, MT=2 outch
// tiling for Cout>=32.
// ---------------------------------------------------------------------------

typedef __attribute__((ext_vector_type(16))) _Float16 v16h;
typedef __attribute__((ext_vector_type(8)))  _Float16 v8h;
typedef __attribute__((ext_vector_type(8)))  float    v8f;

#define BB 1
#define VV 3
#define HH_IMG 512
#define WW_IMG 640
#define DD 48
#define FHc 128
#define FWc 160
#define HHc 64
#define WWc 80
#define NPC (HHc*WWc)      // 5120
#define FNP (FHc*FWc)      // 20480

__device__ __forceinline__ v8f wmma_f16(v16h a, v16h b, v8f c) {
  // 8 args: (neg_a, A, neg_b, B, c_mod, C, reuse_a, reuse_b)
  return __builtin_amdgcn_wmma_f32_16x16x32_f16(false, a, false, b, (short)0, c, false, false);
}

__device__ __forceinline__ v8f zero8() {
  v8f z = {0.f,0.f,0.f,0.f,0.f,0.f,0.f,0.f};
  return z;
}

__device__ __forceinline__ int clampi(int v, int lo, int hi) {
  return v < lo ? lo : (v > hi ? hi : v);
}

// A fragment: lane holds row m = lane&15; K elems: e -> k0+16*half+e.
// wrow is padded f16 (KKp mult of 32) so the two
// 8-half runs are unconditional 16B loads (global_load_b128).
__device__ __forceinline__ v16h load_a(const _Float16* __restrict__ wrow,
                                       int k0, int half) {
  const v8h r0 = *(const v8h*)(wrow + k0 + 16 * half);
  const v8h r1 = *(const v8h*)(wrow + k0 + 16 * half + 8);
  v16h a;
#pragma unroll
  for (int e = 0; e < 8; ++e) { a[e] = r0[e]; a[8 + e] = r1[e]; }
  return a;
}

// B fragment (2D im2col): lane holds column n; 16 consecutive K values
// starting at kb = k0 + 16*half. Fully branchless: flat predicate tree,
// select-based digit carries, clamped always-in-bounds addressing.
template <typename TIn, int KSZ, int STRIDE>
__device__ __forceinline__ v16h gather2d(const TIn* __restrict__ inp,
    int kb, int KK, int Cin, int H, int W, int oy, int ox, bool pixvalid) {
  constexpr int pad = KSZ / 2;
  constexpr int KK2 = KSZ * KSZ;
  int ci = kb / KK2;
  int r = kb - ci * KK2;
  int ky = r / KSZ;
  int kx = r - ky * KSZ;
  const int by = oy * STRIDE - pad;
  const int bx = ox * STRIDE - pad;
  int iy = by + ky, ix = bx + kx;
  v16h b;
#pragma unroll
  for (int e = 0; e < 16; ++e) {
    const bool valid = pixvalid & (kb + e < KK) &
                       (iy >= 0) & (iy < H) & (ix >= 0) & (ix < W);
    const int cic = clampi(ci, 0, Cin - 1);
    const int iyc = clampi(iy, 0, H - 1);
    const int ixc = clampi(ix, 0, W - 1);
    const unsigned id = (unsigned)((cic * H + iyc) * W + ixc);
    const _Float16 v = (_Float16)inp[id];
    b[e] = valid ? v : (_Float16)0;
    // branchless digit-carry advance
    ++kx; ++ix;
    const int cx = (kx == KSZ);
    kx = cx ? 0 : kx;  ix = cx ? bx : ix;
    ky += cx;          iy += cx;
    const int cy = (ky == KSZ);
    ky = cy ? 0 : ky;  iy = cy ? by : iy;
    ci += cy;
  }
  return b;
}

// ---------------------------------------------------------------------------
// 2D conv: one wave computes (16*MT outch) x 32 pixels (2*MT WMMA per step).
// ---------------------------------------------------------------------------
template <typename TIn, int KSZ, int STRIDE, int MT>
__global__ __launch_bounds__(32) void conv2d_wmma_k(
    const TIn* __restrict__ in, const _Float16* __restrict__ wf,
    _Float16* __restrict__ out, int Cin, int Cout, int H, int W, int relu) {
  const int OH = H / STRIDE, OW = W / STRIDE;
  const int NPx = OH * OW;
  const int KK = Cin * KSZ * KSZ;
  const int KKp = (KK + 31) & ~31;
  const int img = blockIdx.z;
  const TIn* inp = in + (size_t)img * Cin * H * W;
  _Float16* outp = out + (size_t)img * Cout * NPx;
  const int n0 = blockIdx.x * 32, m0 = blockIdx.y * 16 * MT;
  const int lane = threadIdx.x;
  const int half = lane >> 4, mrow = lane & 15;

  const int pix0 = n0 + mrow, pix1 = pix0 + 16;
  const bool pv0 = pix0 < NPx, pv1 = pix1 < NPx;
  const int cp0 = pv0 ? pix0 : 0, cp1 = pv1 ? pix1 : 0;
  const int oy0 = cp0 / OW, ox0 = cp0 - oy0 * OW;
  const int oy1 = cp1 / OW, ox1 = cp1 - oy1 * OW;

  const _Float16* wrow[MT];
  v8f acc[MT][2];
#pragma unroll
  for (int mt = 0; mt < MT; ++mt) {
    wrow[mt] = wf + (size_t)(m0 + 16 * mt + mrow) * KKp;
    acc[mt][0] = zero8();
    acc[mt][1] = zero8();
  }

  for (int k0 = 0; k0 < KK; k0 += 32) {
    const int kb = k0 + 16 * half;
    const v16h b0 = gather2d<TIn, KSZ, STRIDE>(inp, kb, KK, Cin, H, W, oy0, ox0, pv0);
    const v16h b1 = gather2d<TIn, KSZ, STRIDE>(inp, kb, KK, Cin, H, W, oy1, ox1, pv1);
#pragma unroll
    for (int mt = 0; mt < MT; ++mt) {
      const v16h a = load_a(wrow[mt], k0, half);
      acc[mt][0] = wmma_f16(a, b0, acc[mt][0]);
      acc[mt][1] = wmma_f16(a, b1, acc[mt][1]);
    }
  }

#pragma unroll
  for (int mt = 0; mt < MT; ++mt)
#pragma unroll
    for (int r = 0; r < 8; ++r) {
      const int M = m0 + 16 * mt + r + 8 * half;
      if (M < Cout) {
        if (pv0) {
          float v = acc[mt][0][r]; if (relu) v = fmaxf(v, 0.f);
          outp[(size_t)M * NPx + pix0] = (_Float16)v;
        }
        if (pv1) {
          float v = acc[mt][1][r]; if (relu) v = fmaxf(v, 0.f);
          outp[(size_t)M * NPx + pix1] = (_Float16)v;
        }
      }
    }
}

// ---------------------------------------------------------------------------
// 3D conv / transposed-3D conv (k=3). Constant 27/9/3 divisors. Optional
// residual (added pre-conv, f16) and per-channel f32 bias.
// ---------------------------------------------------------------------------
template <int DECONV, int HASADD>
__device__ __forceinline__ v16h gather3d(const _Float16* __restrict__ in,
    const _Float16* __restrict__ addin, int kb, int KK, int Cin,
    int Din, int Hin, int Win, int stride,
    int oz, int oy, int ox, bool pixvalid) {
  int ci = kb / 27;
  int r = kb - ci * 27;
  int kz = r / 9; r -= kz * 9;
  int ky = r / 3;
  int kx = r - ky * 3;
  v16h b;
#pragma unroll
  for (int e = 0; e < 16; ++e) {
    bool valid; int iz, iy, ix;
    if (!DECONV) {
      iz = oz * stride + kz - 1;
      iy = oy * stride + ky - 1;
      ix = ox * stride + kx - 1;
      valid = pixvalid & (kb + e < KK) &
              (iz >= 0) & (iz < Din) & (iy >= 0) & (iy < Hin) &
              (ix >= 0) & (ix < Win);
    } else {
      const int pz = oz + kz - 1, py = oy + ky - 1, px = ox + kx - 1;
      iz = pz >> 1; iy = py >> 1; ix = px >> 1;
      valid = pixvalid & (kb + e < KK) &
              (pz >= 0) & (py >= 0) & (px >= 0) &
              ((pz & 1) == 0) & ((py & 1) == 0) & ((px & 1) == 0) &
              (iz < Din) & (iy < Hin) & (ix < Win);
    }
    const int cic = clampi(ci, 0, Cin - 1);
    const int izc = clampi(iz, 0, Din - 1);
    const int iyc = clampi(iy, 0, Hin - 1);
    const int ixc = clampi(ix, 0, Win - 1);
    const unsigned id = (unsigned)(((cic * Din + izc) * Hin + iyc) * Win + ixc);
    _Float16 v = in[id];
    if (HASADD) v = v + addin[id];
    b[e] = valid ? v : (_Float16)0;
    // branchless digit-carry advance
    ++kx;
    const int cx = (kx == 3);
    kx = cx ? 0 : kx;
    ky += cx;
    const int cy = (ky == 3);
    ky = cy ? 0 : ky;
    kz += cy;
    const int cz = (kz == 3);
    kz = cz ? 0 : kz;
    ci += cz;
  }
  return b;
}

template <int DECONV, int HASADD, int MT>
__global__ __launch_bounds__(32) void conv3d_wmma_k(
    const _Float16* __restrict__ in, const _Float16* __restrict__ addin,
    const _Float16* __restrict__ wf, const float* __restrict__ bias,
    _Float16* __restrict__ out, int Cin, int Cout, int Din, int Hin, int Win,
    int stride, int relu) {
  const int Dout = DECONV ? Din * 2 : Din / stride;
  const int Hout = DECONV ? Hin * 2 : Hin / stride;
  const int Wout = DECONV ? Win * 2 : Win / stride;
  const int HWo = Hout * Wout;
  const int NPx = Dout * HWo;
  const int KK = Cin * 27;
  const int KKp = (KK + 31) & ~31;
  const int n0 = blockIdx.x * 32, m0 = blockIdx.y * 16 * MT;
  const int lane = threadIdx.x;
  const int half = lane >> 4, mrow = lane & 15;

  const int pix0 = n0 + mrow, pix1 = pix0 + 16;
  const bool pv0 = pix0 < NPx, pv1 = pix1 < NPx;
  const int cp0 = pv0 ? pix0 : 0, cp1 = pv1 ? pix1 : 0;
  const int oz0 = cp0 / HWo; int r0d = cp0 - oz0 * HWo;
  const int oy0 = r0d / Wout, ox0 = r0d - oy0 * Wout;
  const int oz1 = cp1 / HWo; int r1d = cp1 - oz1 * HWo;
  const int oy1 = r1d / Wout, ox1 = r1d - oy1 * Wout;

  const _Float16* wrow[MT];
  v8f acc[MT][2];
#pragma unroll
  for (int mt = 0; mt < MT; ++mt) {
    wrow[mt] = wf + (size_t)(m0 + 16 * mt + mrow) * KKp;
    acc[mt][0] = zero8();
    acc[mt][1] = zero8();
  }

  for (int k0 = 0; k0 < KK; k0 += 32) {
    const int kb = k0 + 16 * half;
    const v16h b0 = gather3d<DECONV, HASADD>(in, addin, kb, KK, Cin, Din, Hin, Win,
                                             stride, oz0, oy0, ox0, pv0);
    const v16h b1 = gather3d<DECONV, HASADD>(in, addin, kb, KK, Cin, Din, Hin, Win,
                                             stride, oz1, oy1, ox1, pv1);
#pragma unroll
    for (int mt = 0; mt < MT; ++mt) {
      const v16h a = load_a(wrow[mt], k0, half);
      acc[mt][0] = wmma_f16(a, b0, acc[mt][0]);
      acc[mt][1] = wmma_f16(a, b1, acc[mt][1]);
    }
  }

#pragma unroll
  for (int mt = 0; mt < MT; ++mt)
#pragma unroll
    for (int r = 0; r < 8; ++r) {
      const int M = m0 + 16 * mt + r + 8 * half;
      if (M < Cout) {
        const float bs = bias ? bias[M] : 0.f;
        if (pv0) {
          float v = acc[mt][0][r] + bs; if (relu) v = fmaxf(v, 0.f);
          out[(size_t)M * NPx + pix0] = (_Float16)v;
        }
        if (pv1) {
          float v = acc[mt][1][r] + bs; if (relu) v = fmaxf(v, 0.f);
          out[(size_t)M * NPx + pix1] = (_Float16)v;
        }
      }
    }
}

// ---------------------------------------------------------------------------
// Weight repack: f32 [Cout][KK] -> f16 [ceil16(Cout)][ceil32(KK)], zero pad.
// ---------------------------------------------------------------------------
__global__ void repack_w_k(const float* __restrict__ w, _Float16* __restrict__ dst,
                           int Cout, int KK) {
  const int KKp = (KK + 31) & ~31;
  const int Cp = (Cout + 15) & ~15;
  const int t = blockIdx.x * blockDim.x + threadIdx.x;
  if (t >= Cp * KKp) return;
  const int m = t / KKp, k = t - m * KKp;
  dst[t] = (_Float16)((m < Cout && k < KK) ? w[(size_t)m * KK + k] : 0.f);
}

// ---------------------------------------------------------------------------
// Geometry precompute: per view K, R, t, K^-1, R^-1; depth planes; scalars.
// geom layout (floats): view v at v*64: [0..8]=K [9..17]=R [18..20]=t
//                       [21..29]=Kinv [30..38]=Rinv
// [192..239]=depths, [240]=d_start, [241]=d_int
// ---------------------------------------------------------------------------
__device__ void inv3(const float m[9], float* o) {
  const float a=m[0],b=m[1],c=m[2],d=m[3],e=m[4],f=m[5],g=m[6],h=m[7],i=m[8];
  const float A = e*i - f*h, Bv = -(d*i - f*g), Cv = d*h - e*g;
  const float inv = 1.f / (a*A + b*Bv + c*Cv);
  o[0]=A*inv;  o[1]=-(b*i-c*h)*inv; o[2]=(b*f-c*e)*inv;
  o[3]=Bv*inv; o[4]=(a*i-c*g)*inv;  o[5]=-(a*f-c*d)*inv;
  o[6]=Cv*inv; o[7]=-(a*h-b*g)*inv; o[8]=(a*e-b*d)*inv;
}

__global__ void precompute_geom_k(const float* __restrict__ cams,
                                  float* __restrict__ geom) {
  const int v = threadIdx.x;
  if (v < VV) {
    float Km[9], Rm[9];
    float* g = geom + v * 64;
    for (int r = 0; r < 3; ++r)
      for (int c = 0; c < 3; ++c) {
        Rm[r*3+c] = cams[((v*2 + 0)*4 + r)*4 + c];
        Km[r*3+c] = cams[((v*2 + 1)*4 + r)*4 + c];
      }
    for (int k = 0; k < 9; ++k) { g[k] = Km[k]; g[9+k] = Rm[k]; }
    for (int r = 0; r < 3; ++r) g[18+r] = cams[((v*2 + 0)*4 + r)*4 + 3];
    inv3(Km, g + 21);
    inv3(Rm, g + 30);
  }
  if (v == 0) {
    const float ds = cams[((0*2 + 1)*4 + 3)*4 + 0];
    const float di = cams[((0*2 + 1)*4 + 3)*4 + 1];
    geom[240] = ds; geom[241] = di;
    for (int d = 0; d < DD; ++d) geom[192 + d] = ds + (float)d * di;
  }
}

// ---------------------------------------------------------------------------
// Fused cost volume: back-project + per-view bilinear fetch + variance.
// feats f16 in, cv f16 out; all math f32.
// ---------------------------------------------------------------------------
__global__ void cost_volume_k(const _Float16* __restrict__ feats,
                              const float* __restrict__ geom,
                              _Float16* __restrict__ cv, int ref) {
  const int t = blockIdx.x * blockDim.x + threadIdx.x;
  if (t >= DD * NPC) return;
  const int d = t / NPC, p = t - d * NPC;
  const int j = p / WWc, i = p - j * WWc;
  const float px = 2.f * i + 0.5f, py = 2.f * j + 0.5f;

  const float* g = geom + ref * 64;
  const float depth = geom[192 + d];
  const float cx = (g[21]*px + g[22]*py + g[23]) * depth - g[18];
  const float cy = (g[24]*px + g[25]*py + g[26]) * depth - g[19];
  const float cz = (g[27]*px + g[28]*py + g[29]) * depth - g[20];
  const float wx = g[30]*cx + g[31]*cy + g[32]*cz;
  const float wy = g[33]*cx + g[34]*cy + g[35]*cz;
  const float wz = g[36]*cx + g[37]*cy + g[38]*cz;

  int   ridx[VV][4];
  float rwt[VV][4];
  for (int v2 = 0; v2 < VV; ++v2) {
    if (v2 == ref) {
      ridx[v2][0] = (2*j) * FWc + 2*i; rwt[v2][0] = 1.f;
      ridx[v2][1] = ridx[v2][2] = ridx[v2][3] = 0;
      rwt[v2][1] = rwt[v2][2] = rwt[v2][3] = 0.f;
      continue;
    }
    const float* g2 = geom + v2 * 64;
    const float ax = g2[9]*wx  + g2[10]*wy + g2[11]*wz + g2[18];
    const float ay = g2[12]*wx + g2[13]*wy + g2[14]*wz + g2[19];
    const float az = g2[15]*wx + g2[16]*wy + g2[17]*wz + g2[20];
    const float uu = g2[0]*ax + g2[1]*ay + g2[2]*az;
    const float vv = g2[3]*ax + g2[4]*ay + g2[5]*az;
    const float ww = g2[6]*ax + g2[7]*ay + g2[8]*az;
    const float iz = 1.f / (ww + 1e-8f);
    const float u = uu * iz, v = vv * iz;
    const float x0 = floorf(u), y0 = floorf(v);
    int kk = 0;
    for (int dx = 0; dx < 2; ++dx)
      for (int dy = 0; dy < 2; ++dy) {
        const float xi = x0 + dx, yi = y0 + dy;
        const float wgt = (1.f - fabsf(u - xi)) * (1.f - fabsf(v - yi));
        const bool valid = (xi >= 0.f) & (xi <= (float)(FWc-1)) &
                           (yi >= 0.f) & (yi <= (float)(FHc-1));
        const int xic = clampi((int)xi, 0, FWc-1);
        const int yic = clampi((int)yi, 0, FHc-1);
        ridx[v2][kk] = yic * FWc + xic;
        rwt[v2][kk] = valid ? wgt : 0.f;
        ++kk;
      }
  }

  for (int c = 0; c < 32; ++c) {
    float s1 = 0.f, s2 = 0.f;
#pragma unroll
    for (int v2 = 0; v2 < VV; ++v2) {
      const _Float16* fm = feats + ((size_t)v2 * 32 + c) * FNP;
      const float f = rwt[v2][0]*(float)fm[ridx[v2][0]] +
                      rwt[v2][1]*(float)fm[ridx[v2][1]] +
                      rwt[v2][2]*(float)fm[ridx[v2][2]] +
                      rwt[v2][3]*(float)fm[ridx[v2][3]];
      s1 += f; s2 += f * f;
    }
    const float m1 = s1 * (1.f / 3.f);
    cv[((size_t)c * DD + d) * NPC + p] = (_Float16)(s2 * (1.f / 3.f) - m1 * m1);
  }
}

// ---------------------------------------------------------------------------
// Per-pixel softmax over D of -fcv, depth regression, 4-tap prob map.
// ---------------------------------------------------------------------------
__global__ void depth_prob_k(const _Float16* __restrict__ fcv,
                             const float* __restrict__ geom,
                             float* __restrict__ depth,
                             float* __restrict__ prob) {
  const int p = blockIdx.x * blockDim.x + threadIdx.x;
  if (p >= NPC) return;
  float pv[DD];
  float mx = -1e30f;
  for (int d = 0; d < DD; ++d) { pv[d] = -(float)fcv[(size_t)d * NPC + p]; mx = fmaxf(mx, pv[d]); }
  float sum = 0.f;
  for (int d = 0; d < DD; ++d) { pv[d] = expf(pv[d] - mx); sum += pv[d]; }
  const float inv = 1.f / sum;
  float rd = 0.f;
  for (int d = 0; d < DD; ++d) { pv[d] *= inv; rd += geom[192 + d] * pv[d]; }
  depth[p] = rd;
  const float dn = (rd - geom[240]) / geom[241];
  const int l0 = clampi((int)floorf(dn), 0, DD-1);
  const int l1 = clampi(l0 - 1, 0, DD-1);
  const int r0 = clampi((int)ceilf(dn), 0, DD-1);
  const int r1 = clampi(r0 + 1, 0, DD-1);
  prob[p] = pv[l1] + pv[l0] + pv[r0] + pv[r1];
}

// ---------------------------------------------------------------------------
// depth nearest 2x upsample + prob bilinear (half-pixel) 2x upsample.
// ---------------------------------------------------------------------------
__global__ void upsample_k(const float* __restrict__ depth,
                           const float* __restrict__ prob,
                           float* __restrict__ depth_up,
                           float* __restrict__ prob_up) {
  const int t = blockIdx.x * blockDim.x + threadIdx.x;
  if (t >= FNP) return;
  const int y = t / FWc, x = t - y * FWc;
  depth_up[t] = depth[(y >> 1) * WWc + (x >> 1)];
  const float fy = (y + 0.5f) * 0.5f - 0.5f;
  const float fx = (x + 0.5f) * 0.5f - 0.5f;
  const int y0 = (int)floorf(fy), x0 = (int)floorf(fx);
  const float wy = fy - (float)y0, wx = fx - (float)x0;
  const int y0c = clampi(y0, 0, HHc-1), y1c = clampi(y0 + 1, 0, HHc-1);
  const int x0c = clampi(x0, 0, WWc-1), x1c = clampi(x0 + 1, 0, WWc-1);
  prob_up[t] = (1.f-wy)*((1.f-wx)*prob[y0c*WWc+x0c] + wx*prob[y0c*WWc+x1c]) +
               wy      *((1.f-wx)*prob[y1c*WWc+x0c] + wx*prob[y1c*WWc+x1c]);
}

// ---------------------------------------------------------------------------
// Propagation tail: softmax over 9 logits (f16), 3x3 edge-clamped blend.
// ---------------------------------------------------------------------------
__global__ void prop_combine_k(const _Float16* __restrict__ logits9,
                               const float* __restrict__ depth_up,
                               const float* __restrict__ prob_up,
                               float* __restrict__ out) {
  const int t = blockIdx.x * blockDim.x + threadIdx.x;
  if (t >= FNP) return;
  const int y = t / FWc, x = t - y * FWc;
  float l[9], mx = -1e30f;
#pragma unroll
  for (int k = 0; k < 9; ++k) { l[k] = (float)logits9[(size_t)k * FNP + t]; mx = fmaxf(mx, l[k]); }
  float sum = 0.f;
#pragma unroll
  for (int k = 0; k < 9; ++k) { l[k] = expf(l[k] - mx); sum += l[k]; }
  const float inv = 1.f / sum;
  float coarse = 0.f;
#pragma unroll
  for (int dy = 0; dy < 3; ++dy)
#pragma unroll
    for (int dx = 0; dx < 3; ++dx) {
      const int yy = clampi(y + dy - 1, 0, FHc-1);
      const int xx = clampi(x + dx - 1, 0, FWc-1);
      coarse += l[dy*3+dx] * inv * depth_up[yy * FWc + xx];
    }
  out[t] = coarse;
  out[FNP + t] = prob_up[t];
}

// ---------------------------------------------------------------------------
// Host orchestration.
// Flattened input order (recursive insertion order of setup_inputs()):
//  0 imgs, 1 cams,
//  2..9  coarse_img {c0a,c0b,c1a,c1b,c1c,c2a,c2b,c2c}
//  10..21 vol {c1_0,c2_0,c3_0,c0_1,c1_1,c2_1,c3_1,d4_0,d5_0,d6_0,c6_2,c6_2b}
//  22..29 prop_img {c0a..c2c}, 30 prop_head1, 31 prop_head2
// ---------------------------------------------------------------------------
extern "C" void kernel_launch(void* const* d_in, const int* in_sizes, int n_in,
                              void* d_out, int out_size, void* d_ws, size_t ws_size,
                              hipStream_t stream) {
  (void)in_sizes; (void)n_in; (void)out_size; (void)ws_size;

  const float* imgs = (const float*)d_in[0];
  const float* cams = (const float*)d_in[1];
  float* out = (float*)d_out;
  float* ws  = (float*)d_ws;

  // Small f32 regions
  float* geom  = ws;              // 256
  float* depth = ws + 256;        // 5120
  float* prob  = depth + NPC;     // 5120
  float* dup   = prob + NPC;      // 20480
  float* pup   = dup + FNP;       // 20480

  // f16 regions (16B aligned: float offset 51456)
  const size_t SZ_FEATS = (size_t)VV * 32 * FNP;           // 1,966,080 halves
  const size_t SZ_BUF   = (size_t)3 * 8 * HH_IMG * WW_IMG; // 7,864,320 halves
  _Float16* hbase = (_Float16*)(pup + FNP);
  _Float16* feats = hbase;
  _Float16* bufA  = feats + SZ_FEATS;                      // also cost volume
  _Float16* bufB  = bufA + SZ_BUF;                         // also vol scratch
  _Float16* vol0  = bufB + SZ_BUF;                         // c0_1 (8x48x64x80)
  _Float16* fcv   = vol0 + (size_t)8 * DD * NPC;           // 245,760
  _Float16* wfbase = fcv + (size_t)DD * NPC;               // repacked weights

  // vol sub-buffers inside bufB
  _Float16* c1_0 = bufB + 0;
  _Float16* c2_0 = bufB + 500000;
  _Float16* c3_0 = bufB + 650000;
  _Float16* c1_1 = bufB + 700000;
  _Float16* c2_1 = bufB + 1200000;
  _Float16* c3_1 = bufB + 1350000;
  _Float16* c4_0 = bufB + 1400000;
  _Float16* c5_0 = bufB + 1550000;
  _Float16* c6_0 = bufB + 2100000;
  _Float16* fbuf = bufB + 4200000;   // prop feature output (32x128x160)

  // ---- repack all conv weights to padded f16 ----
  const int NL = 29;
  const int lidx[NL]  = {2,3,4,5,6,7,8,9,
                         10,11,12,13,14,15,16,17,18,19,20,
                         22,23,24,25,26,27,28,29, 30,31};
  const int lcout[NL] = {8,8,16,16,16,32,32,32,
                         16,32,64,8,16,32,64,32,16,8,1,
                         8,8,16,16,16,32,32,32, 64,9};
  const int lkk[NL]   = {27,72,200,144,144,400,288,288,
                         864,432,864,864,432,864,1728,1728,864,432,216,
                         27,72,200,144,144,400,288,288, 288,576};
  _Float16* wp16[32];
  {
    size_t off = 0;
    for (int l = 0; l < NL; ++l) {
      const int Cp = (lcout[l] + 15) & ~15;
      const int KKp = (lkk[l] + 31) & ~31;
      wp16[lidx[l]] = wfbase + off;
      const int total = Cp * KKp;
      repack_w_k<<<(total + 255) / 256, 256, 0, stream>>>(
          (const float*)d_in[lidx[l]], wfbase + off, lcout[l], lkk[l]);
      off += (size_t)total;
    }
  }

  // f16-input conv2d dispatch (KSZ/STRIDE/MT templated)
  auto conv2dh = [&](const _Float16* in, const _Float16* wf, _Float16* o, int nImg,
                     int Cin, int Cout, int H, int W, int K, int s, int relu) {
    const int OH = H / s, OW = W / s;
    const int MT = (Cout >= 32) ? 2 : 1;
    dim3 g((OH * OW + 31) / 32, (Cout + 16 * MT - 1) / (16 * MT), nImg);
    if (K == 3) {
      if (MT == 2) conv2d_wmma_k<_Float16,3,1,2><<<g, dim3(32), 0, stream>>>(in, wf, o, Cin, Cout, H, W, relu);
      else         conv2d_wmma_k<_Float16,3,1,1><<<g, dim3(32), 0, stream>>>(in, wf, o, Cin, Cout, H, W, relu);
    } else {
      if (MT == 2) conv2d_wmma_k<_Float16,5,2,2><<<g, dim3(32), 0, stream>>>(in, wf, o, Cin, Cout, H, W, relu);
      else         conv2d_wmma_k<_Float16,5,2,1><<<g, dim3(32), 0, stream>>>(in, wf, o, Cin, Cout, H, W, relu);
    }
  };

  auto conv3dh = [&](const _Float16* in, const _Float16* add, const _Float16* wf,
                     const float* bias, _Float16* o, int Cin, int Cout,
                     int Din, int Hin, int Win, int s, int dec, int relu) {
    const int Do = dec ? Din*2 : Din/s, Ho = dec ? Hin*2 : Hin/s, Wo = dec ? Win*2 : Win/s;
    const int MT = (Cout >= 32) ? 2 : 1;
    dim3 g((Do * Ho * Wo + 31) / 32, (Cout + 16 * MT - 1) / (16 * MT), 1);
    if (dec) {
      if (add) {
        if (MT == 2) conv3d_wmma_k<1,1,2><<<g, dim3(32), 0, stream>>>(in, add, wf, bias, o, Cin, Cout, Din, Hin, Win, s, relu);
        else         conv3d_wmma_k<1,1,1><<<g, dim3(32), 0, stream>>>(in, add, wf, bias, o, Cin, Cout, Din, Hin, Win, s, relu);
      } else {
        if (MT == 2) conv3d_wmma_k<1,0,2><<<g, dim3(32), 0, stream>>>(in, add, wf, bias, o, Cin, Cout, Din, Hin, Win, s, relu);
        else         conv3d_wmma_k<1,0,1><<<g, dim3(32), 0, stream>>>(in, add, wf, bias, o, Cin, Cout, Din, Hin, Win, s, relu);
      }
    } else {
      if (add) {
        if (MT == 2) conv3d_wmma_k<0,1,2><<<g, dim3(32), 0, stream>>>(in, add, wf, bias, o, Cin, Cout, Din, Hin, Win, s, relu);
        else         conv3d_wmma_k<0,1,1><<<g, dim3(32), 0, stream>>>(in, add, wf, bias, o, Cin, Cout, Din, Hin, Win, s, relu);
      } else {
        if (MT == 2) conv3d_wmma_k<0,0,2><<<g, dim3(32), 0, stream>>>(in, add, wf, bias, o, Cin, Cout, Din, Hin, Win, s, relu);
        else         conv3d_wmma_k<0,0,1><<<g, dim3(32), 0, stream>>>(in, add, wf, bias, o, Cin, Cout, Din, Hin, Win, s, relu);
      }
    }
  };

  // image conv chain: first layer reads f32 images, rest f16.
  auto image_conv = [&](const float* img, int nImg, int wb, _Float16* dst) {
    {
      dim3 g((HH_IMG * WW_IMG + 31) / 32, 1, nImg);
      conv2d_wmma_k<float,3,1,1><<<g, dim3(32), 0, stream>>>(
          img, wp16[wb+0], bufA, 3, 8, HH_IMG, WW_IMG, 1);
    }
    conv2dh(bufA, wp16[wb+1], bufB, nImg, 8,  8,  512, 640, 3, 1, 1);
    conv2dh(bufB, wp16[wb+2], bufA, nImg, 8,  16, 512, 640, 5, 2, 1);
    conv2dh(bufA, wp16[wb+3], bufB, nImg, 16, 16, 256, 320, 3, 1, 1);
    conv2dh(bufB, wp16[wb+4], bufA, nImg, 16, 16, 256, 320, 3, 1, 1);
    conv2dh(bufA, wp16[wb+5], bufB, nImg, 16, 32, 256, 320, 5, 2, 1);
    conv2dh(bufB, wp16[wb+6], bufA, nImg, 32, 32, 128, 160, 3, 1, 1);
    conv2dh(bufA, wp16[wb+7], dst,  nImg, 32, 32, 128, 160, 3, 1, 1);
  };

  // Geometry + coarse features (3 images batched)
  precompute_geom_k<<<1, 32, 0, stream>>>(cams, geom);
  image_conv(imgs, VV, 2, feats);

  const float* bias62 = (const float*)d_in[21];

  for (int view = 0; view < VV; ++view) {
    // --- fused cost volume into bufA (32,48,64,80) f16 ---
    cost_volume_k<<<(DD * NPC + 255) / 256, 256, 0, stream>>>(feats, geom, bufA, view);

    // --- 3D U-Net ---
    conv3dh(bufA, nullptr, wp16[13], nullptr, vol0, 32, 8,  48, 64, 80, 1, 0, 1); // c0_1
    conv3dh(bufA, nullptr, wp16[10], nullptr, c1_0, 32, 16, 48, 64, 80, 2, 0, 1);
    conv3dh(c1_0, nullptr, wp16[11], nullptr, c2_0, 16, 32, 24, 32, 40, 2, 0, 1);
    conv3dh(c2_0, nullptr, wp16[12], nullptr, c3_0, 32, 64, 12, 16, 20, 2, 0, 1);
    conv3dh(c1_0, nullptr, wp16[14], nullptr, c1_1, 16, 16, 24, 32, 40, 1, 0, 1);
    conv3dh(c2_0, nullptr, wp16[15], nullptr, c2_1, 32, 32, 12, 16, 20, 1, 0, 1);
    conv3dh(c3_0, nullptr, wp16[16], nullptr, c3_1, 64, 64, 6,  8,  10, 1, 0, 1);
    conv3dh(c3_1, nullptr, wp16[17], nullptr, c4_0, 64, 32, 6,  8,  10, 1, 1, 1); // d4_0
    conv3dh(c4_0, c2_1,    wp16[18], nullptr, c5_0, 32, 16, 12, 16, 20, 1, 1, 1); // d5_0
    conv3dh(c5_0, c1_1,    wp16[19], nullptr, c6_0, 16, 8,  24, 32, 40, 1, 1, 1); // d6_0
    conv3dh(c6_0, vol0,    wp16[20], bias62,  fcv,  8,  1,  48, 64, 80, 1, 0, 0); // c6_2

    // --- softmax / depth regression / prob map / upsample ---
    depth_prob_k<<<(NPC + 255) / 256, 256, 0, stream>>>(fcv, geom, depth, prob);
    upsample_k<<<(FNP + 255) / 256, 256, 0, stream>>>(depth, prob, dup, pup);

    // --- propagation: image conv on this view + heads ---
    image_conv(imgs + (size_t)view * 3 * HH_IMG * WW_IMG, 1, 22, fbuf);
    conv2dh(fbuf, wp16[30], bufA, 1, 32, 64, 128, 160, 3, 1, 1);
    conv2dh(bufA, wp16[31], bufB, 1, 64, 9,  128, 160, 3, 1, 0);

    prop_combine_k<<<(FNP + 255) / 256, 256, 0, stream>>>(
        bufB, dup, pup, out + (size_t)view * 2 * FNP);
  }
}
